// NSM_90872918048839
// MI455X (gfx1250) — matrix operand order repaired and checked by reading, not compiled, not run on hardware
//
#include <hip/hip_runtime.h>
#include <hip/hip_bf16.h>

typedef __attribute__((ext_vector_type(16))) _Float16 v16h;
typedef __attribute__((ext_vector_type(8)))  _Float16 v8h;
typedef __attribute__((ext_vector_type(8)))  float    v8f;

__device__ __forceinline__ float sigf(float x){ return 1.0f/(1.0f+__expf(-x)); }
__device__ __forceinline__ float eluf(float x){ return x > 0.0f ? x : (__expf(x)-1.0f); }

__device__ __forceinline__ v8f wmma_f16(v16h a, v16h b, v8f c){
  return __builtin_amdgcn_wmma_f32_16x16x32_f16(false, a, false, b, (short)0, c, false, false);
}

// ---------------------------------------------------------------------------
// Prep: transpose fp32 [R=256][C=256] (batched) -> f16 [C][R]  (B^T layout)
// ---------------------------------------------------------------------------
__global__ void transpose_h_kernel(const float* __restrict__ src, _Float16* __restrict__ dst, int total){
  int idx = blockIdx.x*256 + threadIdx.x;
  if (idx >= total) return;
  int bt = idx >> 16; int rem = idx & 65535; int r = rem >> 8; int c = rem & 255;
  dst[(size_t)bt*65536 + (size_t)c*256 + r] = (_Float16)src[idx];
}

// concat w_ih|w_hh rows into f16 Bt [1024][512]  (already n-major, no transpose)
__global__ void concat_h_kernel(const float* __restrict__ ih, const float* __restrict__ hh,
                                _Float16* __restrict__ dst){
  int idx = blockIdx.x*256 + threadIdx.x;   // < 1024*512
  int n = idx >> 9, k = idx & 511;
  dst[idx] = (_Float16)(k < 256 ? ih[n*256 + k] : hh[n*256 + (k-256)]);
}

// ---------------------------------------------------------------------------
// Generic WMMA GEMM: D[M,256] = A[M,256] @ Bt^T, A fp32 (lda), Bt f16 [256][256]
// wave computes a 16x64 strip (4 N-tiles), 8 waves / block.
// ---------------------------------------------------------------------------
__global__ void gemm16_f16_kernel(const float* __restrict__ A, int lda,
                                  const _Float16* __restrict__ Bt,
                                  float* __restrict__ D, int ldd, int Mtiles){
  int gwid = (blockIdx.x * blockDim.x + threadIdx.x) >> 5;
  int lane = threadIdx.x & 31;
  int strip = gwid & 3;
  int mt    = gwid >> 2;
  if (mt >= Mtiles) return;
  int mr = lane & 15, hi = lane >> 4;
  const float* Arow = A + (size_t)(mt*16 + mr) * lda;
  v8f acc[4] = {};
  for (int kb = 0; kb < 256; kb += 32){
    v16h a;
    const float* p0 = Arow + kb + hi*16;
    const float* p1 = p0 + 8;
    if (kb + 32 < 256) __builtin_prefetch(Arow + kb + 32, 0, 0);
#pragma unroll
    for (int i = 0; i < 8; ++i){ a[i] = (_Float16)p0[i]; a[8+i] = (_Float16)p1[i]; }
#pragma unroll
    for (int j = 0; j < 4; ++j){
      int n = strip*64 + j*16 + mr;
      v16h bf = *(const v16h*)(Bt + (size_t)n*256 + kb + hi*16);
      acc[j] = wmma_f16(a, bf, acc[j]);
    }
  }
#pragma unroll
  for (int j = 0; j < 4; ++j){
    int n = strip*64 + j*16 + mr;
#pragma unroll
    for (int v = 0; v < 8; ++v){
      int row = mt*16 + v + hi*8;
      D[(size_t)row*ldd + n] = acc[j][v];
    }
  }
}

// ---------------------------------------------------------------------------
// Stage 1 epilogue: logits over NC=6, softmax, V = P5*embd + P[:5]@C[:5]
// one block per row (b*32+q), 256 threads
// ---------------------------------------------------------------------------
__global__ void pv_kernel(const float* __restrict__ t_ws, const float* __restrict__ Cm,
                          const float* __restrict__ embd, float* __restrict__ V){
  __shared__ float parts[256][6];
  __shared__ float P[6];
  int row = blockIdx.x, e = threadIdx.x;
  float tv = t_ws[(size_t)row*256 + e];
#pragma unroll
  for (int c = 0; c < 6; ++c) parts[e][c] = tv * Cm[c*256 + e];
  __syncthreads();
  for (int st = 128; st > 0; st >>= 1){
    if (e < st){
#pragma unroll
      for (int c = 0; c < 6; ++c) parts[e][c] += parts[e+st][c];
    }
    __syncthreads();
  }
  if (e == 0){
    float mx = parts[0][0];
    for (int c = 1; c < 6; ++c) mx = fmaxf(mx, parts[0][c]);
    float ex[6]; float sum = 0.0f;
    for (int c = 0; c < 6; ++c){ ex[c] = __expf(parts[0][c]-mx); sum += ex[c]; }
    float inv = 1.0f/sum;
    for (int c = 0; c < 6; ++c) P[c] = ex[c]*inv;
  }
  __syncthreads();
  float v = P[5]*embd[(size_t)row*256 + e];
#pragma unroll
  for (int c = 0; c < 5; ++c) v += P[c]*Cm[c*256 + e];
  V[(size_t)row*256 + e] = v;
}

// ---------------------------------------------------------------------------
// LSTM (enc & dec): single WG, 512 threads = 16 waves. A=[x|h] 16x512 f16 in LDS,
// Bt f16 [1024][512]. Batch rows 0..7 live in lanes 0..15 of C/D fragments.
// ---------------------------------------------------------------------------
__global__ void __launch_bounds__(512) lstm_kernel(
    const float* __restrict__ X, int xBatchStride, int xStepStride,
    const _Float16* __restrict__ Bt,
    const float* __restrict__ bih, const float* __restrict__ bhh,
    const float* __restrict__ h0, const float* __restrict__ c0,
    int T, float* __restrict__ h_all, float* __restrict__ hT_out, float* __restrict__ cT_out){
  __shared__ alignas(16) _Float16 Ah[16][512];
  __shared__ float hb[8][256];
  __shared__ float cb[8][256];
  __shared__ float gb[8][1024];
  int tid = threadIdx.x;
  int lane = tid & 31, wid = tid >> 5;
  int mr = lane & 15, hi = lane >> 4;
  for (int idx = tid; idx < 8*512; idx += 512) Ah[8 + (idx>>9)][idx & 511] = (_Float16)0.0f;
  for (int idx = tid; idx < 2048; idx += 512){
    int b = idx >> 8, e = idx & 255;
    float h = h0 ? h0[b*256+e] : 0.0f;
    float c = c0 ? c0[b*256+e] : 0.0f;
    hb[b][e] = h; cb[b][e] = c; Ah[b][256+e] = (_Float16)h;
  }
  __syncthreads();
  for (int t = 0; t < T; ++t){
    if (xStepStride != 0 || t == 0){
      for (int idx = tid; idx < 2048; idx += 512){
        int b = idx >> 8, e = idx & 255;
        Ah[b][e] = (_Float16)X[(size_t)b*xBatchStride + (size_t)t*xStepStride + e];
      }
    }
    __syncthreads();
    v8f acc[4] = {};
    for (int kb = 0; kb < 512; kb += 32){
      v16h a;
      {
        const v8h* p0 = (const v8h*)&Ah[mr][kb + hi*16];
        const v8h* p1 = (const v8h*)&Ah[mr][kb + hi*16 + 8];
        v8h lo = *p0, hq = *p1;
#pragma unroll
        for (int i = 0; i < 8; ++i){ a[i] = lo[i]; a[8+i] = hq[i]; }
      }
#pragma unroll
      for (int j = 0; j < 4; ++j){
        int n = wid*64 + j*16 + mr;
        v16h bf = *(const v16h*)(Bt + (size_t)n*512 + kb + hi*16);
        acc[j] = wmma_f16(a, bf, acc[j]);
      }
    }
    if (lane < 16){
#pragma unroll
      for (int j = 0; j < 4; ++j)
#pragma unroll
        for (int v = 0; v < 8; ++v)
          gb[v][wid*64 + j*16 + lane] = acc[j][v];
    }
    __syncthreads();
    if (tid < 256){
      int b = tid >> 5, j0 = (tid & 31)*8;
#pragma unroll
      for (int jj = 0; jj < 8; ++jj){
        int j = j0 + jj;
        float ig = gb[b][j]     + bih[j]     + bhh[j];
        float fg = gb[b][j+256] + bih[j+256] + bhh[j+256];
        float gg = gb[b][j+512] + bih[j+512] + bhh[j+512];
        float og = gb[b][j+768] + bih[j+768] + bhh[j+768];
        float c  = sigf(fg)*cb[b][j] + sigf(ig)*tanhf(gg);
        float h  = sigf(og)*tanhf(c);
        cb[b][j] = c; hb[b][j] = h; Ah[b][256+j] = (_Float16)h;
      }
    }
    __syncthreads();
    if (h_all){
      for (int idx = tid; idx < 2048; idx += 512){
        int b = idx >> 8, e = idx & 255;
        h_all[((size_t)t*8 + b)*256 + e] = hb[b][e];
      }
    }
    __syncthreads();
  }
  if (hT_out){
    for (int idx = tid; idx < 2048; idx += 512){
      int b = idx >> 8, e = idx & 255;
      hT_out[b*256+e] = hb[b][e];
      cT_out[b*256+e] = cb[b][e];
    }
  }
}

// ---------------------------------------------------------------------------
// Attention + r : block per (i,b), i<5 ; scores over q=32, r = attn@V
// ---------------------------------------------------------------------------
__global__ void attn_r_kernel(const float* __restrict__ hdec, const float* __restrict__ V,
                              float* __restrict__ r_ws){
  __shared__ float red[256];
  __shared__ float att[32];
  int i = blockIdx.x >> 3, b = blockIdx.x & 7;
  int tid = threadIdx.x;
  const float* h = hdec + (size_t)(i*8+b)*256;
  int q = tid >> 3, part = tid & 7;
  float s = 0.0f;
  for (int e = part; e < 256; e += 8) s += h[e]*V[((size_t)(b*32+q))*256 + e];
  red[tid] = s; __syncthreads();
  if (part < 4) red[tid] += red[tid+4];
  __syncthreads();
  if (part < 2) red[tid] += red[tid+2];
  __syncthreads();
  if (part == 0) att[q] = red[tid] + red[tid+1];
  __syncthreads();
  if (tid == 0){
    float mx = att[0];
    for (int k = 1; k < 32; ++k) mx = fmaxf(mx, att[k]);
    float sum = 0.0f;
    for (int k = 0; k < 32; ++k){ att[k] = __expf(att[k]-mx); sum += att[k]; }
    float inv = 1.0f/sum;
    for (int k = 0; k < 32; ++k) att[k] *= inv;
  }
  __syncthreads();
  float acc = 0.0f; int e = tid;
  for (int qq = 0; qq < 32; ++qq) acc += att[qq]*V[((size_t)(b*32+qq))*256 + e];
  r_ws[((size_t)(i*8+b))*256 + e] = acc;
}

// ---------------------------------------------------------------------------
// R_i = softmax(D @ r_i) for i=0..4 -> propR[i][b][5], rprime[i][b]; init p=1/128
// ---------------------------------------------------------------------------
__global__ void rsoft_kernel(const float* __restrict__ Dm, const float* __restrict__ r_ws,
                             float* __restrict__ propR, float* __restrict__ rprime,
                             float* __restrict__ p){
  __shared__ float lg[240];
  int tid = threadIdx.x;
  if (tid < 240){
    int i = tid / 48; int rem = tid % 48; int b = rem / 6; int c = rem % 6;
    const float* rr = r_ws + (size_t)(i*8+b)*256;
    const float* dd = Dm + c*256;
    float s = 0.0f;
    for (int e = 0; e < 256; ++e) s += dd[e]*rr[e];
    lg[tid] = s;
  }
  __syncthreads();
  if (tid < 40){
    int i = tid / 8, b = tid % 8;
    const float* l = &lg[i*48 + b*6];
    float mx = l[0];
    for (int c = 1; c < 6; ++c) mx = fmaxf(mx, l[c]);
    float ex[6]; float sum = 0.0f;
    for (int c = 0; c < 6; ++c){ ex[c] = __expf(l[c]-mx); sum += ex[c]; }
    float inv = 1.0f/sum;
    for (int c = 0; c < 5; ++c) propR[(i*8+b)*5 + c] = ex[c]*inv;
    rprime[i*8+b] = ex[5]*inv;
  }
  for (int k = tid; k < 1024; k += 256) p[k] = 1.0f/128.0f;
}

// ---------------------------------------------------------------------------
// Fused EW GEMM + per-step G reduction. Block = one 16-row tile (fixed b,x).
// EW tile computed with WMMA into LDS, never written to HBM.
// G[i,b,x,y] = sum_e elu(EW[b,x,y,e]*r_i[b,e]) * W_r[e],  i = 0..3
// ---------------------------------------------------------------------------
__global__ void __launch_bounds__(256) ew_g_kernel(
    const float* __restrict__ Emat, const _Float16* __restrict__ Wl1t,
    const float* __restrict__ r_ws, const float* __restrict__ Wr,
    float* __restrict__ G){
  __shared__ alignas(16) _Float16 Ah[16][40];
  __shared__ float ew[16][260];
  __shared__ float rb[4][256];
  __shared__ float wrb[256];
  __shared__ float red[16][16][4];
  int tid = threadIdx.x;
  size_t row0 = (size_t)blockIdx.x * 16;
  int b  = (int)(row0 >> 14);
  int x  = (int)((row0 >> 7) & 127);
  int y0 = (int)(row0 & 127);
  {
    wrb[tid & 255] = Wr[tid & 255];
#pragma unroll
    for (int i = 0; i < 4; ++i) rb[i][tid] = r_ws[((size_t)(i*8+b))*256 + tid];
  }
  int lane = tid & 31, wid = tid >> 5;
  int mr = lane & 15, hi = lane >> 4;
  v8f acc[2] = {};
  const float* Abase = Emat + row0*256;
  for (int kb = 0; kb < 256; kb += 32){
    __syncthreads();
    {
      int idx = tid*2; int rr = idx >> 5; int cc = idx & 31;
      Ah[rr][cc]   = (_Float16)Abase[(size_t)rr*256 + kb + cc];
      Ah[rr][cc+1] = (_Float16)Abase[(size_t)rr*256 + kb + cc + 1];
    }
    __syncthreads();
    v16h a;
    {
      const _Float16* p0 = &Ah[mr][hi*16];
      const _Float16* p1 = &Ah[mr][hi*16 + 8];
#pragma unroll
      for (int i = 0; i < 8; ++i){ a[i] = p0[i]; a[8+i] = p1[i]; }
    }
#pragma unroll
    for (int j = 0; j < 2; ++j){
      int n = wid*32 + j*16 + mr;
      v16h bf = *(const v16h*)(Wl1t + (size_t)n*256 + kb + hi*16);
      acc[j] = wmma_f16(a, bf, acc[j]);
    }
  }
#pragma unroll
  for (int j = 0; j < 2; ++j)
#pragma unroll
    for (int v = 0; v < 8; ++v)
      ew[v + hi*8][wid*32 + j*16 + mr] = acc[j][v];
  __syncthreads();
  int rowi = tid >> 4, sub = tid & 15;
  float part[4] = {0.0f, 0.0f, 0.0f, 0.0f};
  for (int c = sub; c < 256; c += 16){
    float v = ew[rowi][c];
#pragma unroll
    for (int i = 0; i < 4; ++i) part[i] += eluf(v*rb[i][c])*wrb[c];
  }
#pragma unroll
  for (int i = 0; i < 4; ++i) red[rowi][sub][i] = part[i];
  __syncthreads();
  if (tid < 64){
    int r2 = tid >> 2, i = tid & 3;
    float s = 0.0f;
#pragma unroll
    for (int k = 0; k < 16; ++k) s += red[r2][k][i];
    G[(((size_t)(i*8+b))*128 + x)*128 + (y0 + r2)] = s;
  }
}

// ---------------------------------------------------------------------------
// p_s logits for all 4 steps (no p dependence): block per (b,n), thread=e
// ---------------------------------------------------------------------------
__global__ void ps_logit_kernel(const float* __restrict__ SW, const float* __restrict__ propR,
                                const float* __restrict__ r_ws, const float* __restrict__ Ws,
                                float* __restrict__ psl){
  __shared__ float red[256][4];
  int bn = blockIdx.x; int b = bn >> 7; int e = threadIdx.x;
  float s[5];
#pragma unroll
  for (int l = 0; l < 5; ++l) s[l] = SW[((size_t)bn*5 + l)*256 + e];
  float w = Ws[e];
#pragma unroll
  for (int i = 0; i < 4; ++i){
    float mix = 0.0f;
#pragma unroll
    for (int l = 0; l < 5; ++l) mix += propR[(i*8+b)*5+l]*s[l];
    red[e][i] = eluf(mix * r_ws[((size_t)(i*8+b))*256 + e]) * w;
  }
  __syncthreads();
  for (int st = 128; st > 0; st >>= 1){
    if (e < st){
#pragma unroll
      for (int i = 0; i < 4; ++i) red[e][i] += red[e+st][i];
    }
    __syncthreads();
  }
  if (e < 4) psl[((size_t)e*8 + b)*128 + (bn & 127)] = red[0][e];
}

// ---------------------------------------------------------------------------
// One propagation step: lp = p@G_i, softmax; ps = softmax(psl_i); blend by rprime
// ---------------------------------------------------------------------------
__global__ void step_kernel(const float* __restrict__ G, const float* __restrict__ psl,
                            const float* __restrict__ rprime, float* __restrict__ p, int i){
  __shared__ float pb[128];
  __shared__ float red[128];
  int b = blockIdx.x, y = threadIdx.x;
  pb[y] = p[b*128 + y];
  __syncthreads();
  const float* g = G + ((size_t)(i*8+b))*128*128;
  float lp = 0.0f;
  for (int x = 0; x < 128; ++x) lp += pb[x]*g[x*128 + y];
  red[y] = lp; __syncthreads();
  for (int st = 64; st > 0; st >>= 1){ if (y < st) red[y] = fmaxf(red[y], red[y+st]); __syncthreads(); }
  float mx = red[0]; __syncthreads();
  float ex = __expf(lp - mx);
  red[y] = ex; __syncthreads();
  for (int st = 64; st > 0; st >>= 1){ if (y < st) red[y] += red[y+st]; __syncthreads(); }
  float pr = ex / red[0]; __syncthreads();
  float ls = psl[((size_t)(i*8+b))*128 + y];
  red[y] = ls; __syncthreads();
  for (int st = 64; st > 0; st >>= 1){ if (y < st) red[y] = fmaxf(red[y], red[y+st]); __syncthreads(); }
  float mx2 = red[0]; __syncthreads();
  float ex2 = __expf(ls - mx2);
  red[y] = ex2; __syncthreads();
  for (int st = 64; st > 0; st >>= 1){ if (y < st) red[y] += red[y+st]; __syncthreads(); }
  float ps = ex2 / red[0];
  float rp = rprime[i*8+b];
  __syncthreads();
  p[b*128 + y] = rp*pr + (1.0f - rp)*ps;
}

// ---------------------------------------------------------------------------
// m = einsum('bn,bl,bnle->be'), x=[m,q], elu MLP -> out.  Block per b.
// ---------------------------------------------------------------------------
__global__ void final_kernel(const float* __restrict__ Sm, const float* __restrict__ p,
                             const float* __restrict__ propR, const float* __restrict__ q,
                             const float* __restrict__ c1w, const float* __restrict__ c1b,
                             const float* __restrict__ c2w, const float* __restrict__ c2b,
                             float* __restrict__ out){
  __shared__ float pb[128];
  __shared__ float prl[5];
  __shared__ float xb[512];
  __shared__ float yb[512];
  int b = blockIdx.x, tid = threadIdx.x;
  if (tid < 128) pb[tid] = p[b*128 + tid];
  if (tid < 5)   prl[tid] = propR[(4*8 + b)*5 + tid];
  __syncthreads();
  if (tid < 256){
    float acc = 0.0f;
    for (int n = 0; n < 128; ++n){
      const float* sr = Sm + ((size_t)(b*128+n)*5)*256 + tid;
      float sm = 0.0f;
#pragma unroll
      for (int l = 0; l < 5; ++l) sm += prl[l]*sr[l*256];
      acc += pb[n]*sm;
    }
    xb[tid] = acc;
  } else {
    xb[tid] = q[b*256 + (tid-256)];
  }
  __syncthreads();
  {
    const float* wrow = c1w + (size_t)tid*512;
    float a = c1b[tid];
    for (int k = 0; k < 512; ++k) a += xb[k]*wrow[k];
    yb[tid] = eluf(a);
  }
  __syncthreads();
  if (tid < 128){
    const float* wrow = c2w + (size_t)tid*512;
    float a = c2b[tid];
    for (int k = 0; k < 512; ++k) a += yb[k]*wrow[k];
    out[b*128 + tid] = a;
  }
}

// ---------------------------------------------------------------------------
extern "C" void kernel_launch(void* const* d_in, const int* in_sizes, int n_in,
                              void* d_out, int out_size, void* d_ws, size_t ws_size,
                              hipStream_t stream){
  (void)in_sizes; (void)n_in; (void)out_size; (void)ws_size;
  const float* embd = (const float*)d_in[0];
  const float* Cm   = (const float*)d_in[1];
  const float* Dm   = (const float*)d_in[2];
  const float* Em   = (const float*)d_in[3];
  const float* Sm   = (const float*)d_in[4];
  // d_in[5] adjacency_mask: unused by the reference computation
  const float* Wm   = (const float*)d_in[6];
  const float* propW= (const float*)d_in[7];
  const float* Wl1  = (const float*)d_in[8];
  const float* Wr   = (const float*)d_in[9];
  const float* Wsv  = (const float*)d_in[10];
  const float* eWih = (const float*)d_in[11];
  const float* eWhh = (const float*)d_in[12];
  const float* eBih = (const float*)d_in[13];
  const float* eBhh = (const float*)d_in[14];
  const float* dWih = (const float*)d_in[15];
  const float* dWhh = (const float*)d_in[16];
  const float* dBih = (const float*)d_in[17];
  const float* dBhh = (const float*)d_in[18];
  const float* c1w  = (const float*)d_in[19];
  const float* c1b  = (const float*)d_in[20];
  const float* c2w  = (const float*)d_in[21];
  const float* c2b  = (const float*)d_in[22];
  float* out = (float*)d_out;

  uintptr_t cur = (uintptr_t)d_ws;
  auto take = [&](size_t bytes) -> void* {
    uintptr_t p = (cur + 255) & ~(uintptr_t)255;
    cur = p + bytes;
    return (void*)p;
  };
  _Float16* Wt    = (_Float16*)take((size_t)65536*2);
  _Float16* Wl1t  = (_Float16*)take((size_t)65536*2);
  _Float16* pWt   = (_Float16*)take((size_t)5*65536*2);
  _Float16* encB  = (_Float16*)take((size_t)1024*512*2);
  _Float16* decB  = (_Float16*)take((size_t)1024*512*2);
  float* t_ws   = (float*)take((size_t)65536*4);
  float* V_ws   = (float*)take((size_t)65536*4);
  float* hT_ws  = (float*)take((size_t)2048*4);
  float* cT_ws  = (float*)take((size_t)2048*4);
  float* hdec   = (float*)take((size_t)10240*4);
  float* r_ws   = (float*)take((size_t)10240*4);
  float* propR  = (float*)take((size_t)256*4);
  float* rprime = (float*)take((size_t)64*4);
  float* p_ws   = (float*)take((size_t)1024*4);
  float* SW_ws  = (float*)take((size_t)1310720*4);
  float* G_ws   = (float*)take((size_t)524288*4);
  float* psl    = (float*)take((size_t)4096*4);

  // weight prep
  transpose_h_kernel<<<256, 256, 0, stream>>>(Wm,   Wt,   65536);
  transpose_h_kernel<<<256, 256, 0, stream>>>(Wl1,  Wl1t, 65536);
  transpose_h_kernel<<<1280,256, 0, stream>>>(propW,pWt,  5*65536);
  concat_h_kernel<<<2048, 256, 0, stream>>>(eWih, eWhh, encB);
  concat_h_kernel<<<2048, 256, 0, stream>>>(dWih, dWhh, decB);

  // stage 1: t = embd @ W ; P/V
  gemm16_f16_kernel<<<8, 256, 0, stream>>>(embd, 256, Wt, t_ws, 256, 16);
  pv_kernel<<<256, 256, 0, stream>>>(t_ws, Cm, embd, V_ws);

  // encoder / decoder LSTMs
  lstm_kernel<<<1, 512, 0, stream>>>(V_ws, 8192, 256, encB, eBih, eBhh,
                                     nullptr, nullptr, 32, nullptr, hT_ws, cT_ws);
  lstm_kernel<<<1, 512, 0, stream>>>(hT_ws, 256, 0, decB, dBih, dBhh,
                                     hT_ws, cT_ws, 5, hdec, nullptr, nullptr);

  // attention reads r[i] for i=0..4 ; per-step softmax mixes ; init p
  attn_r_kernel<<<40, 256, 0, stream>>>(hdec, V_ws, r_ws);
  rsoft_kernel<<<1, 256, 0, stream>>>(Dm, r_ws, propR, rprime, p_ws);

  // SW[b,n,l,:] = S[b,n,l,:] @ propW[l]
  for (int l = 0; l < 5; ++l)
    gemm16_f16_kernel<<<32, 256, 0, stream>>>(Sm + l*256, 1280, pWt + (size_t)l*65536,
                                              SW_ws + l*256, 1280, 64);

  // fused EW GEMM + G precompute (the big one), then p_s logits
  ew_g_kernel<<<8192, 256, 0, stream>>>(Em, Wl1t, r_ws, Wr, G_ws);
  ps_logit_kernel<<<1024, 256, 0, stream>>>(SW_ws, propR, r_ws, Wsv, psl);

  // sequential propagation (tiny)
  for (int i = 0; i < 4; ++i)
    step_kernel<<<8, 128, 0, stream>>>(G_ws, psl, rprime, p_ws, i);

  // m + MLP head
  final_kernel<<<8, 512, 0, stream>>>(Sm, p_ws, propR, hT_ws, c1w, c1b, c2w, c2b, out);
}
